// GatedMultiHeadAttention_69312182223047
// MI455X (gfx1250) — matrix operand order repaired and checked by reading, not compiled, not run on hardware
//
#include <hip/hip_runtime.h>
#include <stdint.h>

#define DEV_INLINE __device__ __forceinline__

typedef __attribute__((ext_vector_type(16))) __bf16 v16bf;
typedef __attribute__((ext_vector_type(8)))  float  v8f;

union FragU {
    uint4 q[2];
    v16bf v;
};

static constexpr int EMBED = 1024;
static constexpr int HEADS = 16;
static constexpr int HDIM  = 64;
static constexpr int BATCH = 2;
static constexpr int SEQ   = 2048;
static constexpr int HD    = HEADS * HDIM;   // 1024
static constexpr int ROWS  = BATCH * SEQ;    // 4096
static constexpr int NQG   = HD + HEADS;     // 1040 (q + gate columns)
static constexpr int NQPAD = 1088;           // 17 * 64, padded for tiling

DEV_INLINE unsigned short f2bf(float f) {
    union { float f; unsigned u; } c; c.f = f;
    unsigned u = c.u;
    unsigned r = u + 0x7FFFu + ((u >> 16) & 1u);  // round-to-nearest-even
    return (unsigned short)(r >> 16);
}

DEV_INLINE float rmax16(float v) {
    v = fmaxf(v, __shfl_xor(v, 8, 16));
    v = fmaxf(v, __shfl_xor(v, 4, 16));
    v = fmaxf(v, __shfl_xor(v, 2, 16));
    v = fmaxf(v, __shfl_xor(v, 1, 16));
    return v;
}

// Per-lane async copy of 32 contiguous bytes global -> LDS (ASYNCcnt path).
// INST_OFFSET applies to both global and LDS addresses (ISA 10.x async ops).
DEV_INLINE void async_copy32(unsigned ldsAddr, const unsigned short* gaddr) {
    asm volatile(
        "global_load_async_to_lds_b128 %0, %1, off\n\t"
        "global_load_async_to_lds_b128 %0, %1, off offset:16"
        :: "v"(ldsAddr), "v"(gaddr) : "memory");
}

// ---------------------------------------------------------------------------
// fp32 -> bf16 elementwise convert
// ---------------------------------------------------------------------------
__global__ void convert_bf16_kernel(const float* __restrict__ x,
                                    unsigned short* __restrict__ y, int n) {
    int i = blockIdx.x * blockDim.x + threadIdx.x;
    if (i < n) y[i] = f2bf(x[i]);
}

// W [K][N] fp32 row-major  ->  WT [Npad][K] bf16 row-major (zero padded rows)
__global__ void convert_transpose_kernel(const float* __restrict__ W,
                                         unsigned short* __restrict__ WT,
                                         int K, int N, int Npad) {
    int i = blockIdx.x * blockDim.x + threadIdx.x;
    int total = Npad * K;
    if (i >= total) return;
    int n = i / K, k = i - n * K;
    float v = (n < N) ? W[(size_t)k * N + n] : 0.0f;
    WT[i] = f2bf(v);
}

// ---------------------------------------------------------------------------
// WMMA GEMM: C[M][N] = A[M][K]_bf16 @ (BT[Npad][K]_bf16)^T + bias
// Each wave: 16 rows x 64 cols. Block: 8 waves -> 128 rows x 64 cols.
// A tiles are staged global->LDS with double-buffered async copies.
// MODE 0: bf16 row-major out (K proj)      MODE 1: q bf16 + gate fp32 split
// MODE 2: V^T store [B][H][D][S] bf16      MODE 3: fp32 out (final proj)
// ---------------------------------------------------------------------------
template <int MODE>
__global__ void gemm_wmma_kernel(const unsigned short* __restrict__ A,
                                 const unsigned short* __restrict__ BT,
                                 const float* __restrict__ bias,
                                 unsigned short* __restrict__ outb,
                                 float* __restrict__ outf,
                                 int N, int K) {
    const int lane  = threadIdx.x & 31;
    const int wave  = threadIdx.x >> 5;
    const int nl    = lane & 15;
    const int half  = lane >> 4;
    const int m_base = blockIdx.y * 128 + wave * 16;
    const int n_base = blockIdx.x * 64;

    // [wave][buffer][lane][32B]: per-lane A-fragment staging, double buffered
    __shared__ __align__(16) unsigned short lds_a[8][2][32][16];

    v8f acc0 = {0,0,0,0,0,0,0,0}, acc1 = {0,0,0,0,0,0,0,0};
    v8f acc2 = {0,0,0,0,0,0,0,0}, acc3 = {0,0,0,0,0,0,0,0};

    // A fragment base: row (m_base+nl), contiguous 16 K per lane-half
    const unsigned short* aRow = A + (size_t)(m_base + nl) * K + half * 16;
    // B fragments: BT row per output col, contiguous 16 K per lane-half
    const unsigned short* b0Row = BT + (size_t)(n_base +  0 + nl) * K + half * 16;
    const unsigned short* b1Row = BT + (size_t)(n_base + 16 + nl) * K + half * 16;
    const unsigned short* b2Row = BT + (size_t)(n_base + 32 + nl) * K + half * 16;
    const unsigned short* b3Row = BT + (size_t)(n_base + 48 + nl) * K + half * 16;

    const unsigned ldsAddr0 = (unsigned)(uintptr_t)&lds_a[wave][0][lane][0];
    const unsigned ldsAddr1 = (unsigned)(uintptr_t)&lds_a[wave][1][lane][0];

    // preload first A tile into buffer 0
    async_copy32(ldsAddr0, aRow);

    for (int k0 = 0; k0 < K; k0 += 32) {
        const int cur = (k0 >> 5) & 1;
        if (k0 + 32 < K) {
            // prefetch next tile into the other buffer, then wait for current
            async_copy32(cur ? ldsAddr0 : ldsAddr1, aRow + k0 + 32);
            asm volatile("s_wait_asynccnt 0x1" ::: "memory");
        } else {
            asm volatile("s_wait_asynccnt 0x0" ::: "memory");
        }
        FragU a;
        const uint4* aL = reinterpret_cast<const uint4*>(&lds_a[wave][cur][lane][0]);
        a.q[0] = aL[0];
        a.q[1] = aL[1];

        FragU b;
        b.q[0] = *reinterpret_cast<const uint4*>(b0Row + k0);
        b.q[1] = *reinterpret_cast<const uint4*>(b0Row + k0 + 8);
        acc0 = __builtin_amdgcn_wmma_f32_16x16x32_bf16(false, a.v, false, b.v,
                                                       (short)0, acc0, false, false);
        b.q[0] = *reinterpret_cast<const uint4*>(b1Row + k0);
        b.q[1] = *reinterpret_cast<const uint4*>(b1Row + k0 + 8);
        acc1 = __builtin_amdgcn_wmma_f32_16x16x32_bf16(false, a.v, false, b.v,
                                                       (short)0, acc1, false, false);
        b.q[0] = *reinterpret_cast<const uint4*>(b2Row + k0);
        b.q[1] = *reinterpret_cast<const uint4*>(b2Row + k0 + 8);
        acc2 = __builtin_amdgcn_wmma_f32_16x16x32_bf16(false, a.v, false, b.v,
                                                       (short)0, acc2, false, false);
        b.q[0] = *reinterpret_cast<const uint4*>(b3Row + k0);
        b.q[1] = *reinterpret_cast<const uint4*>(b3Row + k0 + 8);
        acc3 = __builtin_amdgcn_wmma_f32_16x16x32_bf16(false, a.v, false, b.v,
                                                       (short)0, acc3, false, false);
    }

    float vals[4][8];
#pragma unroll
    for (int j = 0; j < 8; ++j) {
        vals[0][j] = acc0[j]; vals[1][j] = acc1[j];
        vals[2][j] = acc2[j]; vals[3][j] = acc3[j];
    }

#pragma unroll
    for (int g = 0; g < 4; ++g) {
        int col = n_base + g * 16 + nl;
        if (col >= N) continue;
        float bval = bias[col];
#pragma unroll
        for (int j = 0; j < 8; ++j) {
            int row = m_base + j + 8 * half;
            float v = vals[g][j] + bval;
            if (MODE == 0) {
                outb[(size_t)row * HD + col] = f2bf(v);
            } else if (MODE == 1) {
                if (col < HD) outb[(size_t)row * HD + col] = f2bf(v);
                else          outf[(size_t)row * HEADS + (col - HD)] = v;
            } else if (MODE == 2) {
                int b = row >> 11, s = row & (SEQ - 1);
                int h = col >> 6,  d = col & 63;
                outb[(((size_t)b * HEADS + h) * HDIM + d) * SEQ + s] = f2bf(v);
            } else {  // MODE == 3
                outf[(size_t)row * EMBED + col] = v;
            }
        }
    }
}

// ---------------------------------------------------------------------------
// Flash attention w/ online softmax + sigmoid gate.
// One wave = one 16-query tile of one (b,h); 8 waves per block.
// Row-sums of P computed by an extra WMMA against a constant ones-matrix;
// stability max uses a single width-16 reduction per 32-key chunk.
// ---------------------------------------------------------------------------
__global__ void attn_kernel(const unsigned short* __restrict__ q,
                            const unsigned short* __restrict__ k,
                            const unsigned short* __restrict__ vt,
                            const float* __restrict__ gate_logits,
                            const float* __restrict__ mask,
                            unsigned short* __restrict__ attn_out) {
    const int lane = threadIdx.x & 31;
    const int wave = threadIdx.x >> 5;
    const int nl   = lane & 15;
    const int half = lane >> 4;

    const int tile = blockIdx.x * 8 + wave;          // 0..4095
    const int qt = tile & (SEQ / 16 - 1);            // query tile in sequence
    const int bh = tile >> 7;
    const int h  = bh & (HEADS - 1);
    const int b  = bh >> 4;
    const int q_base = qt * 16;

    __shared__ __align__(16) unsigned short plds[8][16][40];

    // Q tile as two A-fragments (d 0..31, d 32..63)
    const unsigned short* qRow = q + (size_t)(b * SEQ + q_base + nl) * HD + h * HDIM + half * 16;
    FragU aQ0, aQ1;
    aQ0.q[0] = *reinterpret_cast<const uint4*>(qRow);
    aQ0.q[1] = *reinterpret_cast<const uint4*>(qRow + 8);
    aQ1.q[0] = *reinterpret_cast<const uint4*>(qRow + 32);
    aQ1.q[1] = *reinterpret_cast<const uint4*>(qRow + 40);

    // constant all-ones B fragment (bf16 1.0 splat) for row-sum WMMA
    FragU ones;
#pragma unroll
    for (int i = 0; i < 2; ++i) {
        ones.q[i].x = 0x3F803F80u; ones.q[i].y = 0x3F803F80u;
        ones.q[i].z = 0x3F803F80u; ones.q[i].w = 0x3F803F80u;
    }

    v8f O0 = {0,0,0,0,0,0,0,0}, O1 = {0,0,0,0,0,0,0,0};
    v8f O2 = {0,0,0,0,0,0,0,0}, O3 = {0,0,0,0,0,0,0,0};
    v8f Lacc = {0,0,0,0,0,0,0,0};            // softmax denominators (row sums)
    float mrun[8];
#pragma unroll
    for (int j = 0; j < 8; ++j) mrun[j] = -1e30f;

    const float scale = 0.125f;  // 1/sqrt(64)
    const float* maskB = mask + (size_t)b * SEQ * SEQ;
    const unsigned short* vBase = vt + ((size_t)(b * HEADS + h) * HDIM) * SEQ;

#pragma unroll 1
    for (int t = 0; t < SEQ; t += 32) {
        // ---- scores: two 16x16 fragments covering keys [t, t+32) ----
        const unsigned short* kRow0 = k + (size_t)(b * SEQ + t + nl) * HD + h * HDIM + half * 16;
        const unsigned short* kRow1 = kRow0 + (size_t)16 * HD;
        if (t + 32 < SEQ) __builtin_prefetch(kRow0 + (size_t)32 * HD, 0, 1);

        FragU bk0, bk1;
        v8f c0 = {0,0,0,0,0,0,0,0}, c1 = {0,0,0,0,0,0,0,0};
        bk0.q[0] = *reinterpret_cast<const uint4*>(kRow0);        // d 0..31
        bk0.q[1] = *reinterpret_cast<const uint4*>(kRow0 + 8);
        bk1.q[0] = *reinterpret_cast<const uint4*>(kRow0 + 32);   // d 32..63
        bk1.q[1] = *reinterpret_cast<const uint4*>(kRow0 + 40);
        c0 = __builtin_amdgcn_wmma_f32_16x16x32_bf16(false, aQ0.v, false, bk0.v,
                                                     (short)0, c0, false, false);
        c0 = __builtin_amdgcn_wmma_f32_16x16x32_bf16(false, aQ1.v, false, bk1.v,
                                                     (short)0, c0, false, false);
        bk0.q[0] = *reinterpret_cast<const uint4*>(kRow1);
        bk0.q[1] = *reinterpret_cast<const uint4*>(kRow1 + 8);
        bk1.q[0] = *reinterpret_cast<const uint4*>(kRow1 + 32);
        bk1.q[1] = *reinterpret_cast<const uint4*>(kRow1 + 40);
        c1 = __builtin_amdgcn_wmma_f32_16x16x32_bf16(false, aQ0.v, false, bk0.v,
                                                     (short)0, c1, false, false);
        c1 = __builtin_amdgcn_wmma_f32_16x16x32_bf16(false, aQ1.v, false, bk1.v,
                                                     (short)0, c1, false, false);

        // ---- scale + mask; half-tile stability max (one reduction) ----
        float s0a[8], s1a[8];
        float mx = -1e30f;
#pragma unroll
        for (int j = 0; j < 8; ++j) {
            int row = q_base + j + 8 * half;
            const float* mrow = maskB + (size_t)row * SEQ + t;
            s0a[j] = c0[j] * scale + mrow[nl];
            s1a[j] = c1[j] * scale + mrow[16 + nl];
            mx = fmaxf(mx, fmaxf(s0a[j], s1a[j]));
        }
        mx = rmax16(mx);  // max over this half-tile (rows of this lane half)

        float alpha[8];
#pragma unroll
        for (int j = 0; j < 8; ++j) {
            float mnew = fmaxf(mrun[j], mx);
            alpha[j] = __expf(mrun[j] - mnew);
            mrun[j] = mnew;
            float p0 = __expf(s0a[j] - mnew);
            float p1 = __expf(s1a[j] - mnew);
            plds[wave][j + 8 * half][nl]      = f2bf(p0);
            plds[wave][j + 8 * half][16 + nl] = f2bf(p1);
        }
#pragma unroll
        for (int j = 0; j < 8; ++j) {
            O0[j] *= alpha[j]; O1[j] *= alpha[j];
            O2[j] *= alpha[j]; O3[j] *= alpha[j];
            Lacc[j] *= alpha[j];
        }

        __builtin_amdgcn_wave_barrier();  // keep LDS write->read ordered (per-wave, in-order DS)

        // P tile (16x32) back as bf16 A-fragment
        FragU aP;
        aP.q[0] = *reinterpret_cast<const uint4*>(&plds[wave][nl][half * 16]);
        aP.q[1] = *reinterpret_cast<const uint4*>(&plds[wave][nl][half * 16 + 8]);

        // row sums of P via WMMA against ones (replaces shuffle reduction)
        Lacc = __builtin_amdgcn_wmma_f32_16x16x32_bf16(false, aP.v, false, ones.v,
                                                       (short)0, Lacc, false, false);

        // ---- accumulate P @ V over 32 keys, 4 d-groups ----
        const unsigned short* vRow = vBase + t + half * 16;
        FragU bvf;
        bvf.q[0] = *reinterpret_cast<const uint4*>(vRow + (size_t)(0  + nl) * SEQ);
        bvf.q[1] = *reinterpret_cast<const uint4*>(vRow + (size_t)(0  + nl) * SEQ + 8);
        O0 = __builtin_amdgcn_wmma_f32_16x16x32_bf16(false, aP.v, false, bvf.v,
                                                     (short)0, O0, false, false);
        bvf.q[0] = *reinterpret_cast<const uint4*>(vRow + (size_t)(16 + nl) * SEQ);
        bvf.q[1] = *reinterpret_cast<const uint4*>(vRow + (size_t)(16 + nl) * SEQ + 8);
        O1 = __builtin_amdgcn_wmma_f32_16x16x32_bf16(false, aP.v, false, bvf.v,
                                                     (short)0, O1, false, false);
        bvf.q[0] = *reinterpret_cast<const uint4*>(vRow + (size_t)(32 + nl) * SEQ);
        bvf.q[1] = *reinterpret_cast<const uint4*>(vRow + (size_t)(32 + nl) * SEQ + 8);
        O2 = __builtin_amdgcn_wmma_f32_16x16x32_bf16(false, aP.v, false, bvf.v,
                                                     (short)0, O2, false, false);
        bvf.q[0] = *reinterpret_cast<const uint4*>(vRow + (size_t)(48 + nl) * SEQ);
        bvf.q[1] = *reinterpret_cast<const uint4*>(vRow + (size_t)(48 + nl) * SEQ + 8);
        O3 = __builtin_amdgcn_wmma_f32_16x16x32_bf16(false, aP.v, false, bvf.v,
                                                     (short)0, O3, false, false);
    }

    // ---- epilogue: 1/l normalize, sigmoid gate, store bf16 ----
#pragma unroll
    for (int j = 0; j < 8; ++j) {
        int row = q_base + j + 8 * half;
        float gl = gate_logits[(size_t)(b * SEQ + row) * HEADS + h];
        float g = 1.0f / (1.0f + __expf(-gl));
        float sc = g / Lacc[j];
        size_t base = (size_t)(b * SEQ + row) * HD + h * HDIM + nl;
        attn_out[base +  0] = f2bf(O0[j] * sc);
        attn_out[base + 16] = f2bf(O1[j] * sc);
        attn_out[base + 32] = f2bf(O2[j] * sc);
        attn_out[base + 48] = f2bf(O3[j] * sc);
    }
}

// ---------------------------------------------------------------------------
extern "C" void kernel_launch(void* const* d_in, const int* in_sizes, int n_in,
                              void* d_out, int out_size, void* d_ws, size_t ws_size,
                              hipStream_t stream) {
    const float* hs   = (const float*)d_in[0];
    const float* mask = (const float*)d_in[1];
    const float* Wq   = (const float*)d_in[2];
    const float* bq   = (const float*)d_in[3];
    const float* Wk   = (const float*)d_in[4];
    const float* bk   = (const float*)d_in[5];
    const float* Wv   = (const float*)d_in[6];
    const float* bvec = (const float*)d_in[7];
    const float* Wo   = (const float*)d_in[8];
    const float* bo   = (const float*)d_in[9];
    float* out = (float*)d_out;
    (void)in_sizes; (void)n_in; (void)out_size; (void)ws_size;

    char* ws = (char*)d_ws;
    size_t off = 0;
    auto alloc = [&](size_t bytes) -> char* {
        char* p = ws + off;
        off += (bytes + 255) & ~size_t(255);
        return p;
    };
    unsigned short* hsb  = (unsigned short*)alloc((size_t)ROWS * EMBED * 2);
    unsigned short* WqT  = (unsigned short*)alloc((size_t)NQPAD * EMBED * 2);
    unsigned short* WkT  = (unsigned short*)alloc((size_t)HD * EMBED * 2);
    unsigned short* WvT  = (unsigned short*)alloc((size_t)HD * EMBED * 2);
    unsigned short* WoT  = (unsigned short*)alloc((size_t)EMBED * HD * 2);
    unsigned short* qb   = (unsigned short*)alloc((size_t)ROWS * HD * 2);
    unsigned short* kb   = (unsigned short*)alloc((size_t)ROWS * HD * 2);
    unsigned short* vtb  = (unsigned short*)alloc((size_t)ROWS * HD * 2);  // [B][H][D][S]
    float*          gate = (float*)         alloc((size_t)ROWS * HEADS * 4);
    unsigned short* atnb = (unsigned short*)alloc((size_t)ROWS * HD * 2);

    const int T = 256;
    // 1) precision conversion + weight transposition
    {
        int n = ROWS * EMBED;
        convert_bf16_kernel<<<(n + T - 1) / T, T, 0, stream>>>(hs, hsb, n);
        int nq = NQPAD * EMBED;
        convert_transpose_kernel<<<(nq + T - 1) / T, T, 0, stream>>>(Wq, WqT, EMBED, NQG, NQPAD);
        int nw = HD * EMBED;
        convert_transpose_kernel<<<(nw + T - 1) / T, T, 0, stream>>>(Wk, WkT, EMBED, HD, HD);
        convert_transpose_kernel<<<(nw + T - 1) / T, T, 0, stream>>>(Wv, WvT, EMBED, HD, HD);
        convert_transpose_kernel<<<(nw + T - 1) / T, T, 0, stream>>>(Wo, WoT, HD, EMBED, EMBED);
    }
    // 2) projections (WMMA): QG, K, V^T
    dim3 blk(256);
    gemm_wmma_kernel<1><<<dim3(NQPAD / 64, ROWS / 128), blk, 0, stream>>>(
        hsb, WqT, bq, qb, gate, NQG, EMBED);
    gemm_wmma_kernel<0><<<dim3(HD / 64, ROWS / 128), blk, 0, stream>>>(
        hsb, WkT, bk, kb, nullptr, HD, EMBED);
    gemm_wmma_kernel<2><<<dim3(HD / 64, ROWS / 128), blk, 0, stream>>>(
        hsb, WvT, bvec, vtb, nullptr, HD, EMBED);
    // 3) gated flash attention (WMMA)
    attn_kernel<<<dim3((BATCH * HEADS * (SEQ / 16)) / 8), blk, 0, stream>>>(
        qb, kb, vtb, gate, mask, atnb);
    // 4) output projection (WMMA) -> fp32 d_out + bias
    gemm_wmma_kernel<3><<<dim3(EMBED / 64, ROWS / 128), blk, 0, stream>>>(
        atnb, WoT, bo, nullptr, out, EMBED, EMBED);
}
